// Encoder_4767413698812
// MI455X (gfx1250) — matrix operand order repaired and checked by reading, not compiled, not run on hardware
//
#include <hip/hip_runtime.h>

#define TOKENS   256
#define TDIM     1280
#define NCODES   8192
#define BATCH    32
#define NROWS    (BATCH * TOKENS)        // 8192 query rows
#define EPB      (TOKENS * TDIM)         // 327680 elements per batch item
#define TOTAL    ((size_t)NROWS * TDIM)  // 10485760 output elements
#define LN_EPS   1e-5f
#define MBLK     64                      // query rows staged in LDS per block
#define NGRP     128                     // codes per group per half (8 N-tiles/wave)
#define KCHUNK   128                     // K elements staged per async chunk (4 wmma steps)

// LDS layout (bytes)
#define LDS_A_OFF     0
#define LDS_A_BYTES   (MBLK * TDIM * 2)                 // 163840
#define LDS_P_OFF     (LDS_A_OFF + LDS_A_BYTES)
#define LDS_P_BYTES   (2 * NGRP * KCHUNK * 2)           // one buffer: 2 halves*128 codes*128k*2B = 65536
#define LDS_V_OFF     (LDS_P_OFF + 2 * LDS_P_BYTES)     // after double buffer
#define LDS_I_OFF     (LDS_V_OFF + MBLK * 32 * 4)
#define LDS_TOTAL     (LDS_I_OFF + MBLK * 32 * 4)       // 311296 B

typedef __attribute__((ext_vector_type(16))) __bf16 v16bf;
typedef __attribute__((ext_vector_type(8)))  __bf16 v8bf;
typedef __attribute__((ext_vector_type(8)))  float  v8f;

__device__ inline v16bf make_v16(v8bf lo, v8bf hi) {
    union { v16bf v; v8bf h[2]; } u;
    u.h[0] = lo; u.h[1] = hi;
    return u.v;
}

// ---------------- Kernel 1: per-batch mean / rsigma (LN over axes (1,2)) ----
__global__ void k_stats(const float* __restrict__ x, float* __restrict__ mu,
                        float* __restrict__ rs, float* __restrict__ lossacc) {
    const int b = blockIdx.x;
    const float* xb = x + (size_t)b * EPB;
    double s = 0.0, ss = 0.0;
    for (int i = threadIdx.x; i < EPB / 4; i += blockDim.x) {
        float4 v = ((const float4*)xb)[i];
        s  += (double)v.x + (double)v.y + (double)v.z + (double)v.w;
        ss += (double)v.x * v.x + (double)v.y * v.y +
              (double)v.z * v.z + (double)v.w * v.w;
    }
    __shared__ double sd[256], sq[256];
    sd[threadIdx.x] = s; sq[threadIdx.x] = ss;
    __syncthreads();
    for (int off = 128; off > 0; off >>= 1) {
        if (threadIdx.x < off) {
            sd[threadIdx.x] += sd[threadIdx.x + off];
            sq[threadIdx.x] += sq[threadIdx.x + off];
        }
        __syncthreads();
    }
    if (threadIdx.x == 0) {
        double mean = sd[0] / (double)EPB;
        double var  = sq[0] / (double)EPB - mean * mean;
        mu[b] = (float)mean;
        rs[b] = rsqrtf((float)var + LN_EPS);
        if (b == 0) lossacc[0] = 0.0f;
    }
}

// ---------------- Kernel 2: normalize + convert queries to bf16 ------------
__global__ void k_norm(const float* __restrict__ x, const float* __restrict__ lnw,
                       const float* __restrict__ lnb, const float* __restrict__ mu,
                       const float* __restrict__ rs, __bf16* __restrict__ fb) {
    const int row = blockIdx.x;            // 0..NROWS-1
    const int b   = row >> 8;              // /TOKENS
    const int tok = row & 255;
    const float m = mu[b], r = rs[b];
    const float* xr = x   + (size_t)row * TDIM;
    const float* wr = lnw + (size_t)tok * TDIM;
    const float* br = lnb + (size_t)tok * TDIM;
    __bf16* o = fb + (size_t)row * TDIM;
    for (int c = threadIdx.x; c < TDIM; c += blockDim.x)
        o[c] = (__bf16)((xr[c] - m) * r * wr[c] + br[c]);
}

// ---------------- Kernel 3: codebook -> bf16 + fp32 row norms --------------
__global__ void k_cb(const float* __restrict__ cb, __bf16* __restrict__ cbb,
                     float* __restrict__ enorm) {
    const int n = blockIdx.x;
    const float* cr = cb + (size_t)n * TDIM;
    __bf16* o = cbb + (size_t)n * TDIM;
    float ss = 0.0f;
    for (int c = threadIdx.x; c < TDIM; c += blockDim.x) {
        float v = cr[c];
        ss += v * v;
        o[c] = (__bf16)v;
    }
    __shared__ float red[256];
    red[threadIdx.x] = ss;
    __syncthreads();
    for (int off = 128; off > 0; off >>= 1) {
        if (threadIdx.x < off) red[threadIdx.x] += red[threadIdx.x + off];
        __syncthreads();
    }
    if (threadIdx.x == 0) enorm[n] = red[0];
}

// ---------------- Kernel 4: WMMA distance + argmin --------------------------
// 256 threads = 8 waves. A: MBLK=64 query rows in LDS (160 KB).
// wave = strip(4) x code-half(2). Code group = 128 codes per half.
// B panels (2 halves x 128 codes x 128 K bf16 = 64 KB) double-buffered in LDS,
// filled with global_load_async_to_lds_b128 one chunk ahead (ASYNCcnt pipeline).
// Per chunk: 32 flattened wmma steps; B fragments prefetched 2 steps ahead and
// pinned before the wmma with sched_barrier(0) so ds_load latency is hidden.

__device__ inline void stage_panel(const __bf16* __restrict__ cbb, int ng, int kc,
                                   unsigned ldsBase, int tid) {
    // 2 halves * 128 codes * (128 K * 2B) = 64 KB = 4096 x 16B transfers; 16/thread
#pragma unroll
    for (int ii = 0; ii < 16; ++ii) {
        const int i    = tid + ii * 256;    // 0..4095
        const int c    = i >> 4;            // packed code row 0..255
        const int k16  = i & 15;            // 16B chunk within 256B row
        const int half = c >> 7;
        const int loc  = c & 127;
        const int codeG = half * (NCODES / 2) + ng + loc;
        unsigned long long g =
            (unsigned long long)(uintptr_t)(cbb + (size_t)codeG * TDIM + kc + k16 * 8);
        unsigned l = ldsBase + (unsigned)(c * (KCHUNK * 2) + k16 * 16);
        asm volatile("global_load_async_to_lds_b128 %0, %1, off"
                     :: "v"(l), "v"(g) : "memory");
    }
}

__launch_bounds__(256)
__global__ void k_argmin(const __bf16* __restrict__ fb, const __bf16* __restrict__ cbb,
                         const float* __restrict__ enorm, int* __restrict__ idx) {
    extern __shared__ __align__(16) char smem[];
    __bf16* fA   = (__bf16*)(smem + LDS_A_OFF);
    float*  sVal = (float*)(smem + LDS_V_OFF);
    int*    sIdx = (int*)(smem + LDS_I_OFF);
    const unsigned panelBase = (unsigned)(uintptr_t)(smem + LDS_P_OFF);

    const int tid     = threadIdx.x;
    const int rowBase = blockIdx.x * MBLK;

    // Kick off async staging of the first B panel, then stage the A tile.
    stage_panel(cbb, 0, 0, panelBase, tid);
    {
        const uint4* src = (const uint4*)(fb + (size_t)rowBase * TDIM);
        uint4* dst = (uint4*)fA;
        const int nvec = MBLK * TDIM / 8;   // 8 bf16 per uint4
        for (int i = tid; i < nvec; i += 256) dst[i] = src[i];
    }
    asm volatile("s_wait_asynccnt 0x0" ::: "memory");
    __syncthreads();

    const int wave  = tid >> 5;
    const int lane  = tid & 31;
    const int l15   = lane & 15;
    const int hi    = lane >> 4;
    const int strip = wave & 3;            // which 16-row strip of A
    const int half  = wave >> 2;           // which 4096-code half
    const int m0    = strip * 16;

    float minv[8];
    int   mini[8];
#pragma unroll
    for (int r = 0; r < 8; ++r) { minv[r] = 3.4e38f; mini[r] = 0; }

    // A fragment base: lane m = m0+l15; K chunks at +hi*16 and +hi*16+8
    const __bf16* aBase = fA + (size_t)(m0 + l15) * TDIM + hi * 16;

    int buf = 0;
    for (int ng = 0; ng < NCODES / 2; ng += NGRP) {
        v8f acc[8];
#pragma unroll
        for (int t = 0; t < 8; ++t) acc[t] = v8f{};

        for (int kc = 0; kc < TDIM; kc += KCHUNK) {
            // stage next chunk into the other buffer (overlaps with compute)
            int kcN = kc + KCHUNK, ngN = ng;
            if (kcN == TDIM) { kcN = 0; ngN = ng + NGRP; }
            if (ngN < NCODES / 2)
                stage_panel(cbb, ngN, kcN, panelBase + (unsigned)((buf ^ 1) * LDS_P_BYTES), tid);

            // ---- compute chunk: 32 flattened wmma steps, s = (k0/32)*8 + t ----
            const __bf16* pan = (const __bf16*)(smem + LDS_P_OFF + buf * LDS_P_BYTES);
            const __bf16* pw  = pan + (size_t)(half * NGRP + l15) * KCHUNK + hi * 16;

            // B fragment address for step s: pw + (s&7)*16*KCHUNK + (s>>3)*32
            v16bf aCur = make_v16(*(const v8bf*)(aBase + kc),
                                  *(const v8bf*)(aBase + kc + 8));
            v16bf aNext = aCur;
            v16bf bCur = make_v16(*(const v8bf*)(pw),
                                  *(const v8bf*)(pw + 8));
            {
                const __bf16* bp = pw + 1 * 16 * KCHUNK;   // step 1
                v16bf t1 = make_v16(*(const v8bf*)bp, *(const v8bf*)(bp + 8));
                bCur = bCur;   // keep order
                // depth-2 pipeline registers
                v16bf bN1 = t1;
#pragma unroll
                for (int s = 0; s < 32; ++s) {
                    const int t  = s & 7;
                    // prefetch B for step s+2
                    v16bf bN2 = bN1;
                    if (s + 2 < 32) {
                        const int s2 = s + 2;
                        const __bf16* bp2 = pw + (size_t)((s2 & 7) * 16) * KCHUNK + (s2 >> 3) * 32;
                        bN2 = make_v16(*(const v8bf*)bp2, *(const v8bf*)(bp2 + 8));
                    }
                    // prefetch A for next k-step, 2 steps before it is needed
                    if (t == 6 && s + 2 < 32) {
                        const int k0n = ((s >> 3) + 1) * 32;
                        aNext = make_v16(*(const v8bf*)(aBase + kc + k0n),
                                         *(const v8bf*)(aBase + kc + k0n + 8));
                    }
                    __builtin_amdgcn_sched_barrier(0);   // pin prefetches above the wmma
                    acc[t] = __builtin_amdgcn_wmma_f32_16x16x32_bf16(
                        false, aCur, false, bCur, (short)0, acc[t], false, false);
                    bCur = bN1;
                    bN1  = bN2;
                    if (t == 7) aCur = aNext;
                }
            }

            // wait for next panel (issued a full compute-chunk ago) + publish
            asm volatile("s_wait_asynccnt 0x0" ::: "memory");
            __syncthreads();
            buf ^= 1;
        }

        // distance = ||e||^2 - 2*dot (||q||^2 is row-constant -> drop for argmin)
#pragma unroll
        for (int t = 0; t < 8; ++t) {
            const int n = half * (NCODES / 2) + ng + t * 16 + l15;
            const float en = enorm[n];
#pragma unroll
            for (int r = 0; r < 8; ++r) {
                float d = en - 2.0f * acc[t][r];
                if (d < minv[r]) { minv[r] = d; mini[r] = n; }  // ascending n keeps first
            }
        }
    }

    // Cross-wave argmin via LDS: row = m0 + r + 8*hi, col = half*16 + l15
#pragma unroll
    for (int r = 0; r < 8; ++r) {
        const int row = m0 + r + 8 * hi;
        const int col = half * 16 + l15;
        sVal[row * 32 + col] = minv[r];
        sIdx[row * 32 + col] = mini[r];
    }
    __syncthreads();
    if (tid < MBLK) {
        float bv = sVal[tid * 32];
        int   bi = sIdx[tid * 32];
        for (int c = 1; c < 32; ++c) {
            float v = sVal[tid * 32 + c];
            int   i2 = sIdx[tid * 32 + c];
            if (v < bv || (v == bv && i2 < bi)) { bv = v; bi = i2; }
        }
        idx[rowBase + tid] = bi;
    }
}

// ---------------- Kernel 5: gather e -> out, accumulate loss ----------------
__global__ void k_out(const float* __restrict__ x, const float* __restrict__ lnw,
                      const float* __restrict__ lnb, const float* __restrict__ cb,
                      const float* __restrict__ mu, const float* __restrict__ rs,
                      const int* __restrict__ idx, float* __restrict__ out,
                      float* __restrict__ lossacc) {
    const size_t stride = (size_t)gridDim.x * blockDim.x;
    float part = 0.0f;
    for (size_t i = (size_t)blockIdx.x * blockDim.x + threadIdx.x; i < TOTAL; i += stride) {
        const int row  = (int)(i / TDIM);
        const int col  = (int)(i - (size_t)row * TDIM);
        const int b    = row >> 8;
        const int widx = (int)(i % (size_t)EPB);
        float f = (x[i] - mu[b]) * rs[b] * lnw[widx] + lnb[widx];
        float e = cb[(size_t)idx[row] * TDIM + col];
        out[i] = e;                      // straight-through forward value
        float d = f - e;
        part += d * d;
    }
    __shared__ float red[256];
    red[threadIdx.x] = part;
    __syncthreads();
    for (int off = 128; off > 0; off >>= 1) {
        if (threadIdx.x < off) red[threadIdx.x] += red[threadIdx.x + off];
        __syncthreads();
    }
    if (threadIdx.x == 0) atomicAdd(lossacc, red[0]);
}

// ---------------- Kernel 6: finalize loss -----------------------------------
__global__ void k_fin(const float* __restrict__ lossacc, float* __restrict__ out) {
    // loss = mean((q-e)^2) + 0.25*mean((e-q)^2) = 1.25 * sum / N
    out[TOTAL] = lossacc[0] * 1.25f / (float)TOTAL;
}

extern "C" void kernel_launch(void* const* d_in, const int* in_sizes, int n_in,
                              void* d_out, int out_size, void* d_ws, size_t ws_size,
                              hipStream_t stream) {
    const float* x   = (const float*)d_in[0];   // [32,256,1280]
    const float* lnw = (const float*)d_in[1];   // [256,1280]
    const float* lnb = (const float*)d_in[2];   // [256,1280]
    const float* cb  = (const float*)d_in[3];   // [8192,1280]
    float* out = (float*)d_out;                 // [32*256*1280] + 1 loss

    char* ws = (char*)d_ws;
    size_t off = 0;
    __bf16* fb  = (__bf16*)(ws + off); off += (size_t)NROWS  * TDIM * 2;  // 20 MB
    __bf16* cbb = (__bf16*)(ws + off); off += (size_t)NCODES * TDIM * 2;  // 20 MB
    float* enorm   = (float*)(ws + off); off += (size_t)NCODES * 4;
    int*   idx     = (int*)  (ws + off); off += (size_t)NCODES * 4;
    float* mu      = (float*)(ws + off); off += (size_t)BATCH * 4;
    float* rs      = (float*)(ws + off); off += (size_t)BATCH * 4;
    float* lossacc = (float*)(ws + off); off += 4;

    k_stats<<<BATCH, 256, 0, stream>>>(x, mu, rs, lossacc);
    k_norm<<<NROWS, 256, 0, stream>>>(x, lnw, lnb, mu, rs, fb);
    k_cb<<<NCODES, 256, 0, stream>>>(cb, cbb, enorm);

    k_argmin<<<NROWS / MBLK, 256, LDS_TOTAL, stream>>>(fb, cbb, enorm, idx);

    k_out<<<1024, 256, 0, stream>>>(x, lnw, lnb, cb, mu, rs, idx, out, lossacc);
    k_fin<<<1, 1, 0, stream>>>(lossacc, out);
}
